// Triformer_10239202033908
// MI455X (gfx1250) — matrix operand order repaired and checked by reading, not compiled, not run on hardware
//
#include <hip/hip_runtime.h>

#define DEV static __device__ __forceinline__

typedef __bf16 bf16;
typedef __attribute__((ext_vector_type(16))) __bf16 v16bf;
typedef __attribute__((ext_vector_type(8)))  float  v8f;
typedef __attribute__((ext_vector_type(8)))  unsigned int v8u;

// ---------- bf16 / math helpers ----------
DEV unsigned short f2bfbits(float f) {
  unsigned u = __builtin_bit_cast(unsigned, f);
  u += 0x7fffu + ((u >> 16) & 1u);            // round-to-nearest-even
  return (unsigned short)(u >> 16);
}
DEV v8f zero8() { v8f z = {0.f,0.f,0.f,0.f,0.f,0.f,0.f,0.f}; return z; }
DEV float rcp_fast(float x) { return __builtin_amdgcn_rcpf(x); }   // raw v_rcp_f32
DEV float fast_tanh(float x) {                 // branch-free: exp + rcp, no div fixup
  float e = __expf(fminf(fmaxf(x, -15.f), 15.f) * 2.f);
  return (e - 1.f) * rcp_fast(e + 1.f);
}
DEV float fast_sigmoid(float x) { return rcp_fast(1.f + __expf(-x)); }

DEV v16bf frag_from(uint4 lo, uint4 hi) {
  v8u u;
  u[0] = lo.x; u[1] = lo.y; u[2] = lo.z; u[3] = lo.w;
  u[4] = hi.x; u[5] = hi.y; u[6] = hi.z; u[7] = hi.w;
  return __builtin_bit_cast(v16bf, u);
}
// A fragment: 16x32 bf16 tile (row-major u16, 16B-aligned base, stride mult of 8):
// lane<16: row=lane, K = 0..15 ; lane>=16: same row, K = 16..31. Two b128 loads.
DEV v16bf load_a_frag(const unsigned short* p, int stride) {
  int lane = threadIdx.x & 31;
  const unsigned short* q = p + (lane & 15)*stride + ((lane >> 4) << 4);
  return frag_from(*(const uint4*)q, *(const uint4*)(q + 8));
}
// B fragment from pre-packed weights: tile = 32 lanes x 16 contiguous bf16.
DEV v16bf load_b_packed(const unsigned short* tile) {
  const unsigned short* q = tile + (threadIdx.x & 31)*16;
  return frag_from(*(const uint4*)q, *(const uint4*)(q + 8));
}
DEV v8f wmma_bf16(v16bf a, v16bf b, v8f c) {
  return __builtin_amdgcn_wmma_f32_16x16x32_bf16(false, a, false, b, (short)0, c, false, false);
}

// ---------- constants ----------
#define BB 8
#define NN 358
#define LAG_ 288
#define HZ 12
#define ROWS (BB*NN)           // 2864 = 179*16

// ---------- weight pack: [K][Nc] f32 -> per-tile lane-contiguous bf16 ----------
// dst[((kt*NT + nt)*32 + khalf*16 + col)*16 + e] = src[(kt*32+khalf*16+e)*Nc + nt*16+col]
__global__ void k_pack_b(const float* __restrict__ s, unsigned short* __restrict__ d,
                         int K, int Nc) {
  int i = blockIdx.x*256 + threadIdx.x;
  if (i >= K*Nc) return;
  int k = i / Nc, n = i - k*Nc;
  int kt = k >> 5, krem = k & 31, khalf = krem >> 4, e = krem & 15;
  int nt = n >> 4, col = n & 15, NT = Nc >> 4;
  d[(((kt*NT + nt)*32) + (khalf << 4) + col)*16 + e] = f2bfbits(s[i]);
}
// Wp2 [512,12] -> packed padded [512,16] (NT=1)
__global__ void k_pack_wp2(const float* __restrict__ s, unsigned short* __restrict__ d) {
  int i = blockIdx.x*256 + threadIdx.x;
  if (i >= 512*16) return;
  int k = i >> 4, n = i & 15;
  float v = (n < 12) ? s[k*12 + n] : 0.f;
  int kt = k >> 5, krem = k & 31, khalf = krem >> 4, e = krem & 15;
  d[(kt*32 + (khalf << 4) + n)*16 + e] = f2bfbits(v);
}

// ---------- hypernetwork: per-node K/V weights (packed bf16) & biases ----------
// wd packed: per (n,w): 8 tiles (kt*4+nt), tile = 32*16 u16 ; bd: f32 [2][N][64]
__global__ void k_hypernet(const float* __restrict__ mem, const float* __restrict__ G1,
                           const float* __restrict__ g1, const float* __restrict__ G2,
                           const float* __restrict__ g2, const float* __restrict__ G3,
                           const float* __restrict__ g3, const float* __restrict__ Bd,
                           const float* __restrict__ Pd, const float* __restrict__ Qd,
                           unsigned short* __restrict__ wd, float* __restrict__ bd) {
  __shared__ float h1[64], h2s[64], m[25], memn[5];
  int n = blockIdx.x, t = threadIdx.x;
  if (t < 5) memn[t] = mem[n*5 + t];
  __syncthreads();
  float a = g1[t];
  for (int j = 0; j < 5; ++j) a += memn[j]*G1[j*64 + t];
  h1[t] = fast_tanh(a);
  __syncthreads();
  float b = g2[t];
  for (int j = 0; j < 64; ++j) b += h1[j]*G2[j*64 + t];
  h2s[t] = fast_tanh(b);
  __syncthreads();
  if (t < 25) {
    float v = g3[t];
    for (int j = 0; j < 64; ++j) v += h2s[j]*G3[j*25 + t];
    m[t] = v;
  }
  __syncthreads();
  for (int w = 0; w < 2; ++w) {
    float v = 0.f;
    for (int mm = 0; mm < 25; ++mm) v += m[mm]*Bd[(w*25 + mm)*64 + t];
    bd[(w*NN + n)*64 + t] = v;
  }
  const int kt = t >> 5, khalf = (t >> 4) & 1, e = t & 15;    // k == t (row of Wd)
  for (int w = 0; w < 2; ++w) {
    float tmp[5];
    for (int p = 0; p < 5; ++p) {
      float v = 0.f;
      for (int mm = 0; mm < 5; ++mm) v += Pd[(w*64 + t)*5 + mm]*m[mm*5 + p];
      tmp[p] = v;
    }
    for (int d = 0; d < 64; ++d) {
      float v = 0.f;
      for (int p = 0; p < 5; ++p) v += tmp[p]*Qd[(w*5 + p)*64 + d];
      int nt = d >> 4, col = d & 15;
      wd[(((n*2 + w)*8 + kt*4 + nt)*32 + (khalf << 4) + col)*16 + e] = f2bfbits(v);
    }
  }
}

// embbase[cut][n][c] = temb[cut,n,:]@We[cut] + be[cut]
__global__ void k_embbase(const float* __restrict__ temb, const float* __restrict__ We,
                          const float* __restrict__ be, float* __restrict__ embb) {
  int cn = blockIdx.x, cut = cn / NN, n = cn - cut*NN, c = threadIdx.x;
  float v = be[cut*64 + c];
  for (int mm = 0; mm < 5; ++mm)
    v += temb[(cut*NN + n)*5 + mm]*We[(cut*5 + mm)*64 + c];
  embb[(cut*NN + n)*64 + c] = v;
}

// ---------- fused gating MLP + t0 (one cut-step) ----------
__global__ void __launch_bounds__(256)
k_gating_t0(const float* __restrict__ prevx, int bstride,
            const unsigned short* __restrict__ W1a, const float* __restrict__ b1a,
            const unsigned short* __restrict__ W1b, const float* __restrict__ b1b,
            const unsigned short* __restrict__ W2a, const float* __restrict__ b2a,
            const unsigned short* __restrict__ W2b, const float* __restrict__ b2b,
            const float* __restrict__ embc, float* __restrict__ t0, int first) {
  __shared__ __align__(16) unsigned short aT[16*64];
  __shared__ __align__(16) unsigned short hbuf[16*1024];
  __shared__ float red[4*16*16];
  __shared__ float gbuf[2*16*64];
  const int tid = threadIdx.x, wave = tid >> 5;
  const int lane = tid & 31, colL = lane & 15, mb = (lane >> 4) << 3;
  const int row0 = blockIdx.x * 16;

  if (first) {                       // cut 0: prev carry zero -> t0 = embbase
    for (int idx = tid; idx < 16*64; idx += 256) {
      int r = row0 + (idx >> 6), c = idx & 63;
      int b = r / NN, n = r - b*NN;
      t0[(b*NN + n)*64 + c] = embc[n*64 + c];
    }
    return;
  }

  for (int idx = tid; idx < 16*64; idx += 256) {
    int r = row0 + (idx >> 6), c = idx & 63;
    int b = r / NN, n = r - b*NN;
    aT[idx] = f2bfbits(prevx[(long)b*bstride + n*64 + c]);
  }
  __syncthreads();
  v16bf a0 = load_a_frag(aT, 64);
  v16bf a1 = load_a_frag(aT + 32, 64);
  const int nt2 = wave & 3, khalf = wave >> 2;

  for (int br = 0; br < 2; ++br) {
    const unsigned short* Wa = br ? W2a : W1a;   // packed, NT=256, KT=2
    const float*          ba = br ? b2a : b1a;
    const unsigned short* Wb = br ? W2b : W1b;   // packed, NT=4,  KT=128
    const float*          bb = br ? b2b : b1b;
    v8f acc2 = zero8();
    for (int qtr = 0; qtr < 4; ++qtr) {
      if (lane == 0)
        __builtin_prefetch(Wb + ((qtr*32 + khalf*16)*4 + nt2)*512, 0, 1);
      for (int tl = 0; tl < 8; ++tl) {           // GEMM1: hidden quarter
        int ntloc = wave*8 + tl;
        int ng    = qtr*64 + ntloc;              // global n-tile
        v8f acc = zero8();
        acc = wmma_bf16(a0, load_b_packed(Wa + (0*256 + ng)*512), acc);
        acc = wmma_bf16(a1, load_b_packed(Wa + (1*256 + ng)*512), acc);
#pragma unroll
        for (int i = 0; i < 8; ++i)
          hbuf[(mb + i)*1024 + ntloc*16 + colL] =
              f2bfbits(fast_tanh(acc[i] + ba[ng*16 + colL]));
      }
      __syncthreads();
      for (int ks = 0; ks < 16; ++ks) {          // GEMM2 partial over this quarter
        int kk = khalf*512 + ks*32;
        int kt = qtr*32 + khalf*16 + ks;
        acc2 = wmma_bf16(load_a_frag(hbuf + kk, 1024),
                         load_b_packed(Wb + (kt*4 + nt2)*512), acc2);
      }
      __syncthreads();
    }
    if (khalf == 1) {
#pragma unroll
      for (int i = 0; i < 8; ++i) red[nt2*256 + (mb + i)*16 + colL] = acc2[i];
    }
    __syncthreads();
    if (khalf == 0) {
#pragma unroll
      for (int i = 0; i < 8; ++i)
        gbuf[br*1024 + (mb + i)*64 + nt2*16 + colL] =
            acc2[i] + red[nt2*256 + (mb + i)*16 + colL] + bb[nt2*16 + colL];
    }
    __syncthreads();
  }
  for (int idx = tid; idx < 16*64; idx += 256) {
    int m = idx >> 6, c = idx & 63;
    int r = row0 + m, b = r / NN, n = r - b*NN;
    float g = fast_tanh(gbuf[m*64 + c]) * fast_sigmoid(gbuf[1024 + m*64 + c]);
    t0[(b*NN + n)*64 + c] = embc[n*64 + c] + g;
  }
}

// ---------- fused per-node K/V WMMA projection + attention + output MLP ----------
__global__ void __launch_bounds__(256)
k_kv_attn(const float* __restrict__ xin, int Lin, int cut, int psz, int cuts,
          const float* __restrict__ t0, const unsigned short* __restrict__ wd,
          const float* __restrict__ bd, const float* __restrict__ Ps,
          const float* __restrict__ Bs, const float* __restrict__ hist,
          const float* __restrict__ Wst, const float* __restrict__ bst,
          int layer0, float* __restrict__ xnext) {
  __shared__ __align__(16) unsigned short tbf[80*64];
  __shared__ float kvb[2*80*64];
  __shared__ float qf[8*64];
  __shared__ float obuf[8*64];
  __shared__ float o2[8*64];
  const int n = blockIdx.x, tid = threadIdx.x, wave = tid >> 5;
  const int lane = tid & 31, colL = lane & 15, mb = (lane >> 4) << 3;
  const int T = psz + 1, BT = BB*T, Mt = (BT + 15) >> 4, Mpad = Mt << 4;

  for (int idx = tid; idx < Mpad*64; idx += 256) {
    int r = idx >> 6, c = idx & 63;
    float v = 0.f;
    if (r < BT) {
      int b = r / T, tt = r - b*T;
      if (tt == 0) { v = t0[(b*NN + n)*64 + c]; qf[b*64 + c] = v; }
      else {
        int l = cut*psz + tt - 1;
        if (layer0) {                            // fold start projection (IN_DIM=3)
          const float* hp = hist + (((long)b*LAG_ + l)*NN + n)*3;
          v = bst[c] + hp[0]*Wst[c] + hp[1]*Wst[64 + c] + hp[2]*Wst[128 + c];
        } else {
          v = xin[(((long)b*Lin + l)*NN + n)*64 + c];
        }
      }
    }
    tbf[idx] = f2bfbits(v);
  }
  __syncthreads();

  const int total = 2*Mt*4;                      // K & V projections
  for (int tix = wave; tix < total; tix += 8) {
    int w = tix / (Mt*4), rem = tix - w*(Mt*4);
    int mt = rem >> 2, nt = rem & 3;
    v8f acc = zero8();
    v16bf aa0 = load_a_frag(tbf + mt*16*64, 64);
    v16bf aa1 = load_a_frag(tbf + mt*16*64 + 32, 64);
    const unsigned short* wp = wd + ((n*2 + w)*8)*512;   // 8 packed tiles (L2)
    acc = wmma_bf16(aa0, load_b_packed(wp + (0*4 + nt)*512), acc);
    acc = wmma_bf16(aa1, load_b_packed(wp + (1*4 + nt)*512), acc);
#pragma unroll
    for (int i = 0; i < 8; ++i)
      kvb[w*80*64 + (mt*16 + mb + i)*64 + nt*16 + colL] =
          acc[i] + bd[(w*NN + n)*64 + nt*16 + colL];
  }
  __syncthreads();

  if (tid < 64) {                                // (b,head) softmax attention, T<=9
    int b = tid >> 3, hd = tid & 7;
    float sc[9], mx = -1e30f;
    for (int t2 = 0; t2 < T; ++t2) {
      float s = 0.f;
      for (int j = 0; j < 8; ++j)
        s += qf[b*64 + hd*8 + j] * kvb[(b*T + t2)*64 + hd*8 + j];
      s *= 0.35355339059327373f;                 // 1/sqrt(8)
      sc[t2] = s; mx = fmaxf(mx, s);
    }
    float den = 0.f;
    for (int t2 = 0; t2 < T; ++t2) { sc[t2] = __expf(sc[t2] - mx); den += sc[t2]; }
    float inv = rcp_fast(den);
    for (int j = 0; j < 8; ++j) {
      float o = 0.f;
      for (int t2 = 0; t2 < T; ++t2)
        o += sc[t2] * kvb[80*64 + (b*T + t2)*64 + hd*8 + j];
      obuf[b*64 + hd*8 + j] = o * inv;
    }
  }
  __syncthreads();

  for (int idx = tid; idx < 512; idx += 256) {   // o2 = tanh(o@Ps0 + Bs0)
    int b = idx >> 6, c = idx & 63;
    float v = Bs[c];
    for (int j = 0; j < 64; ++j) v += obuf[b*64 + j]*Ps[j*64 + c];
    o2[idx] = fast_tanh(v);
  }
  __syncthreads();
  for (int idx = tid; idx < 512; idx += 256) {   // out = o2@Ps1 + Bs1
    int b = idx >> 6, c = idx & 63;
    float v = Bs[64 + c];
    for (int j = 0; j < 64; ++j) v += o2[b*64 + j]*Ps[(64 + j)*64 + c];
    xnext[(((long)b*cuts + cut)*NN + n)*64 + c] = v;
  }
}

// ---------- skip accumulation: skip += reshape(xnext)@Wskip + bskip ----------
__global__ void __launch_bounds__(256)
k_skip(const float* __restrict__ xnext, const unsigned short* __restrict__ Wskip,
       const float* __restrict__ bskip, float* __restrict__ skip, int cuts, int init) {
  __shared__ __align__(16) unsigned short abuf[16*512];
  const int tid = threadIdx.x, wave = tid >> 5;
  const int lane = tid & 31, colL = lane & 15, mb = (lane >> 4) << 3;
  const int row0 = blockIdx.x * 16;
  const int K = cuts * 64;
  const int nt0 = wave*2, nt1 = wave*2 + 1;
  v8f acc0 = zero8(), acc1 = zero8();
  for (int k0 = 0; k0 < K; k0 += 512) {
    int kc = (K - k0 < 512) ? (K - k0) : 512;
    __syncthreads();
    for (int idx = tid; idx < 16*kc; idx += 256) {
      int m = idx / kc, kk = idx - m*kc, k = k0 + kk;
      int r = row0 + m, b = r / NN, n = r - b*NN;
      int cut = k >> 6, c = k & 63;
      abuf[m*512 + kk] = f2bfbits(xnext[(((long)b*cuts + cut)*NN + n)*64 + c]);
    }
    __syncthreads();
    for (int kk = 0; kk < kc; kk += 32) {
      int kt = (k0 + kk) >> 5;
      v16bf a = load_a_frag(abuf + kk, 512);
      acc0 = wmma_bf16(a, load_b_packed(Wskip + (kt*16 + nt0)*512), acc0);
      acc1 = wmma_bf16(a, load_b_packed(Wskip + (kt*16 + nt1)*512), acc1);
    }
  }
#pragma unroll
  for (int i = 0; i < 8; ++i) {
    int r = row0 + mb + i;
    float v0 = acc0[i] + bskip[nt0*16 + colL];
    float v1 = acc1[i] + bskip[nt1*16 + colL];
    float* p0 = &skip[(long)r*256 + nt0*16 + colL];
    float* p1 = &skip[(long)r*256 + nt1*16 + colL];
    *p0 = init ? v0 : (*p0 + v0);
    *p1 = init ? v1 : (*p1 + v1);
  }
}

// ---------- head: relu(skip) @ Wp1 relu @ Wp2 -> transpose [B,HZ,N,1] ----------
__global__ void __launch_bounds__(256)
k_head(const float* __restrict__ skip, const unsigned short* __restrict__ Wp1,
       const float* __restrict__ bp1, const unsigned short* __restrict__ Wp2p,
       const float* __restrict__ bp2, float* __restrict__ out) {
  __shared__ __align__(16) unsigned short a1[16*256];
  __shared__ __align__(16) unsigned short h2[16*512];
  __shared__ float red[8*16*16];
  const int tid = threadIdx.x, wave = tid >> 5;
  const int lane = tid & 31, colL = lane & 15, mb = (lane >> 4) << 3;
  const int row0 = blockIdx.x * 16;
  for (int idx = tid; idx < 16*256; idx += 256) {
    int r = row0 + (idx >> 8);
    a1[idx] = f2bfbits(fmaxf(skip[(long)r*256 + (idx & 255)], 0.f));
  }
  __syncthreads();
  for (int q = 0; q < 4; ++q) {                  // GEMM1: [16,256]@[256,512], NT=32
    int nt = wave*4 + q, ncol = nt*16;
    v8f acc = zero8();
    for (int k = 0; k < 256; k += 32)
      acc = wmma_bf16(load_a_frag(a1 + k, 256),
                      load_b_packed(Wp1 + ((k >> 5)*32 + nt)*512), acc);
#pragma unroll
    for (int i = 0; i < 8; ++i)
      h2[(mb + i)*512 + ncol + colL] = f2bfbits(fmaxf(acc[i] + bp1[ncol + colL], 0.f));
  }
  __syncthreads();
  v8f acc = zero8();                             // GEMM2: [16,512]@[512,16pad], NT=1
  for (int ks = 0; ks < 2; ++ks) {
    int k = wave*64 + ks*32;
    acc = wmma_bf16(load_a_frag(h2 + k, 512), load_b_packed(Wp2p + (k >> 5)*512), acc);
  }
#pragma unroll
  for (int i = 0; i < 8; ++i) red[wave*256 + (mb + i)*16 + colL] = acc[i];
  __syncthreads();
  for (int idx = tid; idx < 256; idx += 256) {
    int m = idx >> 4, c = idx & 15;
    if (c < 12) {
      int r = row0 + m, b = r / NN, node = r - b*NN;
      float v = bp2[c];
      for (int w = 0; w < 8; ++w) v += red[w*256 + m*16 + c];
      out[((long)b*HZ + c)*NN + node] = v;
    }
  }
}

// ================= host side =================
extern "C" void kernel_launch(void* const* d_in, const int* in_sizes, int n_in,
                              void* d_out, int out_size, void* d_ws, size_t ws_size,
                              hipStream_t stream) {
  (void)in_sizes; (void)n_in; (void)out_size; (void)ws_size;
  const float* hist = (const float*)d_in[0];
  // top-level params (jax sorts dict keys): W_start, Wp1, Wp2, b_start, bp1, bp2, layers...
  const float* W_start = (const float*)d_in[5];
  const float* Wp1     = (const float*)d_in[6];
  const float* Wp2     = (const float*)d_in[7];
  const float* b_start = (const float*)d_in[8];
  const float* bp1     = (const float*)d_in[9];
  const float* bp2     = (const float*)d_in[10];
  // per-layer leaves, alphabetical:
  // 0 Bd,1 Bs,2 G1,3 G2,4 G3,5 Pd,6 Ps,7 Qd,8 W1a,9 W1b,10 W2a,11 W2b,12 We,
  // 13 Wskip,14 b1a,15 b1b,16 b2a,17 b2b,18 be,19 bskip,20 g1,21 g2,22 g3,23 mem,24 temb
  const int cuts_a[4] = {36, 9, 3, 1};
  const int psz_a[4]  = {8, 4, 3, 3};
  const float* P[4][25];
  for (int l = 0; l < 4; ++l)
    for (int k = 0; k < 25; ++k) P[l][k] = (const float*)d_in[11 + l*25 + k];

  char* ws = (char*)d_ws;
  size_t off = 0;
  auto alloc = [&](size_t bytes) -> void* {
    void* p = ws + off; off = (off + bytes + 255) & ~(size_t)255; return p;
  };
  unsigned short *w1a_b[4], *w1b_b[4], *w2a_b[4], *w2b_b[4], *wskip_b[4];
  for (int l = 0; l < 4; ++l) {
    w1a_b[l] = (unsigned short*)alloc(64*4096*2);
    w1b_b[l] = (unsigned short*)alloc(4096*64*2);
    w2a_b[l] = (unsigned short*)alloc(64*4096*2);
    w2b_b[l] = (unsigned short*)alloc(4096*64*2);
    wskip_b[l] = (unsigned short*)alloc((size_t)cuts_a[l]*64*256*2);
  }
  unsigned short* wp1_b = (unsigned short*)alloc(256*512*2);
  unsigned short* wp2_b = (unsigned short*)alloc(512*16*2);
  unsigned short* wd_b  = (unsigned short*)alloc((size_t)NN*2*64*64*2);
  float* bd_ws = (float*)alloc((size_t)2*NN*64*4);
  float* embb  = (float*)alloc((size_t)36*NN*64*4);
  float* t0    = (float*)alloc((size_t)BB*NN*64*4);
  float* skip  = (float*)alloc((size_t)ROWS*256*4);
  float* xbuf[4];
  for (int l = 0; l < 4; ++l)
    xbuf[l] = (float*)alloc((size_t)BB*cuts_a[l]*NN*64*4);

  auto packB = [&](const float* s, unsigned short* d, int K, int Nc) {
    int cnt = K*Nc;
    k_pack_b<<<(cnt + 255)/256, 256, 0, stream>>>(s, d, K, Nc);
  };
  for (int l = 0; l < 4; ++l) {
    packB(P[l][8],  w1a_b[l], 64, 4096);
    packB(P[l][9],  w1b_b[l], 4096, 64);
    packB(P[l][10], w2a_b[l], 64, 4096);
    packB(P[l][11], w2b_b[l], 4096, 64);
    packB(P[l][13], wskip_b[l], cuts_a[l]*64, 256);
  }
  packB(Wp1, wp1_b, 256, 512);
  k_pack_wp2<<<(512*16 + 255)/256, 256, 0, stream>>>(Wp2, wp2_b);

  const float* xin = nullptr;
  int Lin = LAG_;
  for (int l = 0; l < 4; ++l) {
    const int cuts = cuts_a[l], psz = psz_a[l];
    k_hypernet<<<NN, 64, 0, stream>>>(P[l][23], P[l][2], P[l][20], P[l][3], P[l][21],
                                      P[l][4], P[l][22], P[l][0], P[l][5], P[l][7],
                                      wd_b, bd_ws);
    k_embbase<<<cuts*NN, 64, 0, stream>>>(P[l][24], P[l][12], P[l][18], embb);
    for (int cut = 0; cut < cuts; ++cut) {
      const float* prevx = xbuf[l] + (size_t)(cut > 0 ? cut - 1 : 0)*NN*64;
      k_gating_t0<<<ROWS/16, 256, 0, stream>>>(
          prevx, cuts*NN*64,
          w1a_b[l], P[l][14], w1b_b[l], P[l][15],
          w2a_b[l], P[l][16], w2b_b[l], P[l][17],
          embb + (size_t)cut*NN*64, t0, cut == 0 ? 1 : 0);
      k_kv_attn<<<NN, 256, 0, stream>>>(
          xin, Lin, cut, psz, cuts, t0, wd_b, bd_ws,
          P[l][6], P[l][1], hist, W_start, b_start,
          l == 0 ? 1 : 0, xbuf[l]);
    }
    k_skip<<<ROWS/16, 256, 0, stream>>>(xbuf[l], wskip_b[l], P[l][19], skip,
                                        cuts, l == 0 ? 1 : 0);
    xin = xbuf[l];
    Lin = cuts;
  }
  k_head<<<ROWS/16, 256, 0, stream>>>(skip, wp1_b, bp1, wp2_b, bp2, (float*)d_out);
}
